// MsDL_CPU_62586263438030
// MI455X (gfx1250) — matrix operand order repaired and checked by reading, not compiled, not run on hardware
//
#include <hip/hip_runtime.h>

typedef __attribute__((ext_vector_type(16))) _Float16 v16h;
typedef __attribute__((ext_vector_type(8)))  float    v8f;

#define B_SZ 64
#define L_SZ 2048
#define U_SZ 64
#define K_SZ 32
#define LEAKY 0.1f

// ---------------- gfx1250 async / tensor data movement helpers ----------------

// Async global->LDS copy, 16B per lane. VDST = LDS byte offset (low 32 bits of the
// generic shared-memory address), VADDR = 64-bit global address. Tracked by ASYNCcnt.
#define ASYNC_B128(gsrc, ldst)                                                   \
  asm volatile("global_load_async_to_lds_b128 %0, %1, off"                       \
               :: "v"((unsigned)(unsigned long long)(ldst)),                     \
                  "v"((unsigned long long)(gsrc)) : "memory")

// Wait until at most n async copies from this wave remain outstanding.
#define WAIT_ASYNC_LE(n) asm volatile("s_wait_asynccnt " #n ::: "memory")

#if __has_builtin(__builtin_amdgcn_s_wait_tensorcnt)
#define WAIT_TENSOR() __builtin_amdgcn_s_wait_tensorcnt(0)
#else
#define WAIT_TENSOR() asm volatile("s_wait_tensorcnt 0" ::: "memory")
#endif

#if __has_builtin(__builtin_amdgcn_tensor_load_to_lds)
#define HAVE_TDM 1
typedef unsigned int v4u  __attribute__((ext_vector_type(4)));
typedef int          v4i_t __attribute__((ext_vector_type(4)));
typedef int          v8i_t __attribute__((ext_vector_type(8)));

// Build a D# (ISA 8.3/8.4) for a contiguous row-major 2D f32 tile and issue the TDM.
__device__ __forceinline__ void tdm_load_2d_f32(const void* gsrc, void* lds_dst,
                                                unsigned rows, unsigned cols) {
  unsigned long long ga = (unsigned long long)gsrc;
  unsigned lds = (unsigned)(unsigned long long)lds_dst;  // LDS aperture: offset in [31:0]
  v4u g0;
  g0.x = 1u;                                              // count=1, user mode
  g0.y = lds;                                             // lds_addr
  g0.z = (unsigned)ga;                                    // global_addr[31:0]
  g0.w = (unsigned)((ga >> 32) & 0x01ffffffu) | (2u << 30);  // addr[56:32] | type=2
  v8i_t g1;
  g1[0] = (int)(2u << 16);                                // mask=0, data_size=2 (4B)
  g1[1] = (int)((cols & 0xffffu) << 16);                  // tensor_dim0[15:0] @bit48
  g1[2] = (int)(((cols >> 16) & 0xffffu) | ((rows & 0xffffu) << 16));  // dim0 hi | dim1 lo
  g1[3] = (int)(((rows >> 16) & 0xffffu) | ((cols & 0xffffu) << 16));  // dim1 hi | tile_dim0
  g1[4] = (int)(rows & 0xffffu);                          // tile_dim1 | tile_dim2=0
  g1[5] = (int)cols;                                      // tensor_dim0_stride[31:0]
  g1[6] = 0;
  g1[7] = 0;
  v4i_t z4 = {0, 0, 0, 0};
#if __has_include(<hip/amd_detail/amd_gfx1250_TDM.h>)
  v8i_t z8 = {0, 0, 0, 0, 0, 0, 0, 0};
  __builtin_amdgcn_tensor_load_to_lds(g0, g1, z4, z4, z8, 0);
#else
  __builtin_amdgcn_tensor_load_to_lds(g0, g1, z4, z4, 0);
#endif
}
#endif  // HAVE_TDM

// ---------------- WMMA helpers ----------------

__device__ __forceinline__ v8f wmma16(const v16h a, const v16h b, v8f c) {
  return __builtin_amdgcn_wmma_f32_16x16x32_f16(false, a, false, b, (short)0, c, false, false);
}

// A fragment: row-major A[m][k], leading dim ld (halves).
__device__ __forceinline__ v16h load_a(const _Float16* A, int ld, int mrow0, int kc, int lane) {
  v16h a;
  const _Float16* p = A + (mrow0 + (lane & 15)) * ld + kc + ((lane >> 4) * 8);
#pragma unroll
  for (int i = 0; i < 8; ++i) { a[i] = p[i]; a[8 + i] = p[16 + i]; }
  return a;
}

// B fragment from transposed storage BT[n][k], leading dim ld.
__device__ __forceinline__ v16h load_b(const _Float16* BT, int ld, int ncol0, int kc, int lane) {
  v16h b;
  const _Float16* p = BT + (ncol0 + (lane & 15)) * ld + kc + ((lane >> 4) * 8);
#pragma unroll
  for (int i = 0; i < 8; ++i) { b[i] = p[i]; b[8 + i] = p[16 + i]; }
  return b;
}

// ---------------- xt = x @ W_in ----------------
__global__ __launch_bounds__(256) void k_xt(const float* __restrict__ x,
                                            const float* __restrict__ Win,
                                            float* __restrict__ xt) {
  __shared__ float    Wstage[64 * 64];
  __shared__ float    Xstage[32 * 64];
  __shared__ _Float16 Ax[32 * 64];
  __shared__ _Float16 WT[64 * 64];
  int tid = threadIdx.x, lane = tid & 31, wave = tid >> 5;
  long row0 = (long)blockIdx.x * 32;
  const float* gx = x + row0 * 64;
  ASYNC_B128(gx + tid * 4, &Xstage[tid * 4]);
  ASYNC_B128(gx + (tid + 256) * 4, &Xstage[(tid + 256) * 4]);
#ifdef HAVE_TDM
  if (wave == 0) { tdm_load_2d_f32(Win, Wstage, 64, 64); WAIT_TENSOR(); }
#else
  for (int i = tid; i < 64 * 64; i += 256) Wstage[i] = Win[i];
#endif
  WAIT_ASYNC_LE(0);
  __syncthreads();
  for (int i = tid; i < 64 * 64; i += 256) {
    int k = i >> 6, n = i & 63;
    WT[n * 64 + k] = (_Float16)Wstage[k * 64 + n];
  }
  for (int i = tid; i < 32 * 64; i += 256) Ax[i] = (_Float16)Xstage[i];
  __syncthreads();
  int mt = wave >> 2, nt = wave & 3;
  v8f acc = {};
  acc = wmma16(load_a(Ax, 64, mt * 16, 0, lane),  load_b(WT, 64, nt * 16, 0, lane),  acc);
  acc = wmma16(load_a(Ax, 64, mt * 16, 32, lane), load_b(WT, 64, nt * 16, 32, lane), acc);
  int n = nt * 16 + (lane & 15);
  int mhi = (lane >> 4) * 8;
#pragma unroll
  for (int r = 0; r < 8; ++r)
    xt[(row0 + mt * 16 + mhi + r) * 64 + n] = acc[r];
}

// ---------------- recurrence: one block per batch, double-buffered async input ----------------
__global__ __launch_bounds__(256) void k_reservoir(const float* __restrict__ xt,
                                                   const float* __restrict__ Wres,
                                                   const float* __restrict__ radii,
                                                   int skip, float* __restrict__ H) {
  __shared__ float    Wstage[64 * 64];
  __shared__ _Float16 WT[64 * 64];     // WT[n][k] = W[k][n]
  __shared__ float    hf[32 * 64];     // state f32
  __shared__ _Float16 hh[32 * 64];     // state f16 (WMMA A operand)
  __shared__ float    xbuf[2 * 2048];  // ping-pong async-staged input blocks
  int tid = threadIdx.x, lane = tid & 31, wave = tid >> 5;
  int b = blockIdx.x;
  int s = skip + 1;
  int T = L_SZ - s;
  int J = (T + s - 1) / s;
  float radius = radii[skip];
  const float* W = Wres + (long)skip * 64 * 64;
  const float* xtb = xt + (long)b * L_SZ * 64;
  float* Hb = H + (long)b * T * 64;
  int nseg = s * 16;  // 16B segments per input block (s*64 floats, contiguous)
#ifdef HAVE_TDM
  if (wave == 0) { tdm_load_2d_f32(W, Wstage, 64, 64); WAIT_TENSOR(); }
#else
  for (int i = tid; i < 64 * 64; i += 256) Wstage[i] = W[i];
#endif
  for (int i = tid; i < 32 * 64; i += 256) { hf[i] = 0.f; hh[i] = (_Float16)0.f; }
  // stage block 0
  if (tid < nseg)       ASYNC_B128(xtb + tid * 4, &xbuf[tid * 4]);
  if (tid + 256 < nseg) ASYNC_B128(xtb + (tid + 256) * 4, &xbuf[(tid + 256) * 4]);
  WAIT_ASYNC_LE(0);
  __syncthreads();
  for (int i = tid; i < 64 * 64; i += 256) {
    int k = i >> 6, n = i & 63;
    WT[n * 64 + k] = (_Float16)Wstage[k * 64 + n];
  }
  __syncthreads();
  int mt = wave >> 2, nt = wave & 3;
  int n = nt * 16 + (lane & 15);
  int mhi = (lane >> 4) * 8;
  for (int j = 0; j < J; ++j) {
    // overlap: stage block j+1 while computing step j
    if (j + 1 < J) {
      const float* gsrc = xtb + (long)(j + 1) * s * 64;
      float* dst = &xbuf[((j + 1) & 1) * 2048];
      if (tid < nseg)       ASYNC_B128(gsrc + tid * 4, dst + tid * 4);
      if (tid + 256 < nseg) ASYNC_B128(gsrc + (tid + 256) * 4, dst + (tid + 256) * 4);
    }
    v8f acc = {};
    acc = wmma16(load_a(hh, 64, mt * 16, 0, lane),  load_b(WT, 64, nt * 16, 0, lane),  acc);
    acc = wmma16(load_a(hh, 64, mt * 16, 32, lane), load_b(WT, 64, nt * 16, 32, lane), acc);
    const float* xb = &xbuf[(j & 1) * 2048];
    float hnew[8];
#pragma unroll
    for (int r = 0; r < 8; ++r) {
      int i = mt * 16 + mhi + r;
      long t = (long)j * s + i;
      float xv = (i < s && t < T) ? xb[i * 64 + n] : 0.f;  // pad region stays 0
      float hold = hf[i * 64 + n];
      hnew[r] = LEAKY * hold + (1.0f - LEAKY) * tanhf(xv + acc[r] * radius);
    }
    __syncthreads();
#pragma unroll
    for (int r = 0; r < 8; ++r) {
      int i = mt * 16 + mhi + r;
      if (i < s) {
        long t = (long)j * s + i;
        hf[i * 64 + n] = hnew[r];
        hh[i * 64 + n] = (_Float16)hnew[r];
        if (t < T) Hb[t * 64 + n] = hnew[r];
      }
    }
    WAIT_ASYNC_LE(0);  // block j+1 landed (overlapped with this whole step)
    __syncthreads();
  }
}

// ---------------- HtH, HtX, colsum(H), sum(X): double-buffered async streaming ----------------
__global__ __launch_bounds__(256) void k_gram(const float* __restrict__ H,
                                              const float* __restrict__ x,
                                              int skip,
                                              float* __restrict__ HtH,
                                              float* __restrict__ HtX,
                                              float* __restrict__ colsumH,
                                              float* __restrict__ sumX) {
  __shared__ float    Hc32[2 * 2048];
  __shared__ float    Xc32[2 * 2048];
  __shared__ _Float16 HcT[64 * 32];  // HcT[u][t]
  __shared__ _Float16 XcT[64 * 32];  // XcT[d][t]
  __shared__ float red[256];
  int tid = threadIdx.x, lane = tid & 31, wave = tid >> 5;
  int b = blockIdx.x;
  int s = skip + 1, T = L_SZ - s;
  const float* Hb = H + (long)b * T * 64;
  const float* Xb = x + ((long)b * L_SZ + s) * 64;
  int mt = wave >> 1;
  int nt0 = (wave & 1) * 2, nt1 = nt0 + 1;
  v8f accG0 = {}, accG1 = {}, accX0 = {}, accX1 = {};
  float csum = 0.f, xsum = 0.f;
  int u = tid & 63;
  int jr = tid >> 6;  // 0..3
  int NC = (T + 31) >> 5;

  auto stage_chunk = [&](int c) -> bool {  // returns true if staged via async
    long t0 = (long)c * 32;
    float* hd = &Hc32[(c & 1) * 2048];
    float* xd = &Xc32[(c & 1) * 2048];
    if (t0 + 32 <= (long)T) {
      const float* gH = Hb + t0 * 64;
      const float* gX = Xb + t0 * 64;
      ASYNC_B128(gH + tid * 4,         hd + tid * 4);
      ASYNC_B128(gH + (tid + 256) * 4, hd + (tid + 256) * 4);
      ASYNC_B128(gX + tid * 4,         xd + tid * 4);
      ASYNC_B128(gX + (tid + 256) * 4, xd + (tid + 256) * 4);
      return true;
    }
    for (int e = tid; e < 2048; e += 256) {
      int j = e >> 6, uu = e & 63;
      long t = t0 + j;
      hd[e] = (t < T) ? Hb[t * 64 + uu] : 0.f;
      xd[e] = (t < T) ? Xb[t * 64 + uu] : 0.f;
    }
    return false;
  };

  stage_chunk(0);
  for (int c = 0; c < NC; ++c) {
    if (c + 2 < NC) __builtin_prefetch(Hb + ((long)(c + 2) * 32) * 64 + tid, 0, 1);
    bool nextAsync = false;
    if (c + 1 < NC) nextAsync = stage_chunk(c + 1);
    // 4 async instrs per wave per chunk, in-order completion: <=4 means chunk c landed
    if (nextAsync) { WAIT_ASYNC_LE(4); } else { WAIT_ASYNC_LE(0); }
    __syncthreads();
    const float* hsrc = &Hc32[(c & 1) * 2048];
    const float* xsrc = &Xc32[(c & 1) * 2048];
#pragma unroll
    for (int i = 0; i < 8; ++i) {
      int j = jr + 4 * i;
      float hv = hsrc[j * 64 + u];
      float xv = xsrc[j * 64 + u];
      HcT[u * 32 + j] = (_Float16)hv;
      XcT[u * 32 + j] = (_Float16)xv;
      csum += hv;
      xsum += xv;
    }
    __syncthreads();
    v16h aH = load_a(HcT, 32, mt * 16, 0, lane);
    accG0 = wmma16(aH, load_b(HcT, 32, nt0 * 16, 0, lane), accG0);
    accG1 = wmma16(aH, load_b(HcT, 32, nt1 * 16, 0, lane), accG1);
    accX0 = wmma16(aH, load_b(XcT, 32, nt0 * 16, 0, lane), accX0);
    accX1 = wmma16(aH, load_b(XcT, 32, nt1 * 16, 0, lane), accX1);
    __syncthreads();
  }
  int mhi = (lane >> 4) * 8;
  int ncl = lane & 15;
  float* Gb = HtH + (size_t)b * 4096;
  float* Xo = HtX + (size_t)b * 4096;
#pragma unroll
  for (int r = 0; r < 8; ++r) {
    int m = mt * 16 + mhi + r;
    Gb[m * 64 + nt0 * 16 + ncl] = accG0[r];
    Gb[m * 64 + nt1 * 16 + ncl] = accG1[r];
    Xo[m * 64 + nt0 * 16 + ncl] = accX0[r];
    Xo[m * 64 + nt1 * 16 + ncl] = accX1[r];
  }
  red[tid] = csum;
  __syncthreads();
  if (tid < 64) colsumH[b * 64 + tid] = red[tid] + red[tid + 64] + red[tid + 128] + red[tid + 192];
  __syncthreads();
  red[tid] = xsum;
  __syncthreads();
  if (tid == 0) {
    float tt = 0.f;
    for (int i = 0; i < 256; ++i) tt += red[i];
    sumX[b] = tt;
  }
}

// ---------------- (HtH + I) W = HtX, per batch; write features ----------------
__global__ __launch_bounds__(64) void k_solve(const float* __restrict__ HtH,
                                              const float* __restrict__ HtX,
                                              int skip, float* __restrict__ out) {
  __shared__ float A[64][65];
  __shared__ float Bm[64][64];
  int tid = threadIdx.x;
  int b = blockIdx.x;
  const float* Ab = HtH + (size_t)b * 4096;
  const float* Xb = HtX + (size_t)b * 4096;
  for (int i = 0; i < 64; ++i) {
    A[i][tid] = Ab[i * 64 + tid] + ((i == tid) ? 1.0f : 0.0f);  // REGULAR = 1
    Bm[i][tid] = Xb[i * 64 + tid];
  }
  __syncthreads();
  for (int k = 0; k < 63; ++k) {  // forward elimination (SPD, no pivot)
    if (tid > k) {
      float f = A[tid][k] / A[k][k];
      for (int j = k; j < 64; ++j) A[tid][j] -= f * A[k][j];
      for (int d = 0; d < 64; ++d) Bm[tid][d] -= f * Bm[k][d];
    }
    __syncthreads();
  }
  for (int k = 63; k >= 0; --k) {  // back-sub: thread owns column tid
    float v = Bm[k][tid];
    for (int j = k + 1; j < 64; ++j) v -= A[k][j] * Bm[j][tid];
    Bm[k][tid] = v / A[k][k];
  }
  __syncthreads();
  float* fb = out + (((long)b * K_SZ + skip) * 64) * 64;
  for (int u = 0; u < 64; ++u) fb[u * 64 + tid] = Bm[u][tid];
}

// ---------------- fns[skip] = sum_b( colsumH_b . rowsum(Wout_b) - sumX_b ) ----------------
__global__ __launch_bounds__(64) void k_fns(const float* __restrict__ colsumH,
                                            const float* __restrict__ sumX,
                                            const float* __restrict__ features,
                                            int skip, float* __restrict__ fout) {
  __shared__ float red[64];
  int tid = threadIdx.x;  // batch index
  const float* fb = features + (((long)tid * K_SZ + skip) * 64) * 64;
  const float* cs = colsumH + tid * 64;
  float acc = 0.f;
  for (int u = 0; u < 64; ++u) {
    float rw = 0.f;
    for (int d = 0; d < 64; ++d) rw += fb[u * 64 + d];
    acc += cs[u] * rw;
  }
  acc -= sumX[tid];
  red[tid] = acc;
  __syncthreads();
  if (tid == 0) {
    float tot = 0.f;
    for (int i = 0; i < 64; ++i) tot += red[i];
    fout[skip] = tot;
  }
}

extern "C" void kernel_launch(void* const* d_in, const int* in_sizes, int n_in,
                              void* d_out, int out_size, void* d_ws, size_t ws_size,
                              hipStream_t stream) {
  const float* x     = (const float*)d_in[0];
  const float* radii = (const float*)d_in[1];
  const float* Win   = (const float*)d_in[2];
  const float* Wres  = (const float*)d_in[3];
  // d_in[4] = regular (identity * 1.0) — baked into the solver.
  float* out = (float*)d_out;

  float* xt      = (float*)d_ws;                       // B*L*U
  float* H       = xt + (size_t)B_SZ * L_SZ * U_SZ;    // B*L*U (upper bound)
  float* HtH     = H + (size_t)B_SZ * L_SZ * U_SZ;     // B*64*64
  float* HtX     = HtH + (size_t)B_SZ * 64 * 64;       // B*64*64
  float* colsumH = HtX + (size_t)B_SZ * 64 * 64;       // B*64
  float* sumX    = colsumH + (size_t)B_SZ * 64;        // B
  float* fnorms  = out + (size_t)B_SZ * K_SZ * 64 * 64;

  k_xt<<<(B_SZ * L_SZ) / 32, 256, 0, stream>>>(x, Win, xt);
  for (int skip = 0; skip < K_SZ; ++skip) {
    k_reservoir<<<B_SZ, 256, 0, stream>>>(xt, Wres, radii, skip, H);
    k_gram<<<B_SZ, 256, 0, stream>>>(H, x, skip, HtH, HtX, colsumH, sumX);
    k_solve<<<B_SZ, 64, 0, stream>>>(HtH, HtX, skip, out);
    k_fns<<<1, 64, 0, stream>>>(colsumH, sumX, out, skip, fnorms);
  }
}
